// GNN_Edge_31550829756485
// MI455X (gfx1250) — matrix operand order repaired and checked by reading, not compiled, not run on hardware
//
#include <hip/hip_runtime.h>
#include <hip/hip_bf16.h>

#define N_NODES 100000
#define K_NB    16
#define C_IN    128
#define C_H     256
#define C_OUTD  128
#define BN_EPS  1e-5f
#define NEG_SLOPE 0.2f
#define M_TILES (N_NODES / 16)   // 6250 row tiles of 16

typedef __bf16 bf16_2  __attribute__((ext_vector_type(2)));
typedef __bf16 bf16_4  __attribute__((ext_vector_type(4)));
typedef __bf16 bf16_8  __attribute__((ext_vector_type(8)));
typedef __bf16 bf16_16 __attribute__((ext_vector_type(16)));
typedef float  f32_4   __attribute__((ext_vector_type(4)));
typedef float  f32_8   __attribute__((ext_vector_type(8)));

// ---------------------------------------------------------------------------
// async 16B global -> LDS copy (CDNA5 GLOBAL_LOAD_ASYNC_TO_LDS_B128, ASYNCcnt)
// with synchronous fallback if the builtin is unavailable on this toolchain.
// ---------------------------------------------------------------------------
#if defined(__AMDGCN__) && __has_builtin(__builtin_amdgcn_global_load_async_to_lds_b128)
#define HAVE_ASYNC_LDS 1
typedef int v4i_t __attribute__((vector_size(16)));
typedef __attribute__((address_space(1))) v4i_t* gv4p_t;
typedef __attribute__((address_space(3))) v4i_t* lv4p_t;
#else
#define HAVE_ASYNC_LDS 0
#endif

__device__ __forceinline__ void copy16_to_lds(const void* g, void* l) {
#if HAVE_ASYNC_LDS
  __builtin_amdgcn_global_load_async_to_lds_b128((gv4p_t)g, (lv4p_t)l, 0, 0);
#else
  *(uint4*)l = *(const uint4*)g;
#endif
}

__device__ __forceinline__ void wait_lds_copies() {
#if HAVE_ASYNC_LDS
#if __has_builtin(__builtin_amdgcn_s_wait_asynccnt)
  __builtin_amdgcn_s_wait_asynccnt(0);
#else
  asm volatile("s_wait_asynccnt 0" ::: "memory");
#endif
#endif
}

// ---------------------------------------------------------------------------
// prep: convert W1 -> bf16 (layer-1 consumes raw input, no BN fold), copy b1,
// zero all stat accumulators (6 * C_H floats, contiguous).
// ---------------------------------------------------------------------------
__global__ __launch_bounds__(256) void prep_kernel(
    const float* __restrict__ W1, const float* __restrict__ b1,
    __bf16* __restrict__ W1f, float* __restrict__ b1f,
    float* __restrict__ stats)
{
  const int i = blockIdx.x * blockDim.x + threadIdx.x;
  const int total = C_H * C_IN;
  for (int idx = i; idx < total; idx += gridDim.x * blockDim.x)
    W1f[idx] = (__bf16)W1[idx];
  if (i < C_H) b1f[i] = b1[i];
  if (i < 6 * C_H) stats[i] = 0.0f;
}

// ---------------------------------------------------------------------------
// convert input features f32 -> bf16
// ---------------------------------------------------------------------------
__global__ __launch_bounds__(256) void convert_x_kernel(
    const float* __restrict__ X, __bf16* __restrict__ Xb, int total)
{
  const int i = blockIdx.x * blockDim.x + threadIdx.x;
  for (int idx = i; idx < total; idx += gridDim.x * blockDim.x)
    Xb[idx] = (__bf16)X[idx];
}

// ---------------------------------------------------------------------------
// fold: from layer-l stats compute a = g*rsqrt(var+eps), c = beta - mean*a,
// then fold into next layer: Wf[j,k] = bf16(W[j,k]*a[k]),
// bf_out[j] = b[j] + sum_k W[j,k]*c[k].   One block of 256 threads.
// ---------------------------------------------------------------------------
__global__ __launch_bounds__(256) void fold_kernel(
    const float* __restrict__ Wn, const float* __restrict__ bn,
    const float* __restrict__ g,  const float* __restrict__ beta,
    const float* __restrict__ sum, const float* __restrict__ sumsq,
    float invN, __bf16* __restrict__ Wf, float* __restrict__ bf_out,
    int rows, int ck)
{
  __shared__ float a_s[C_H];
  __shared__ float c_s[C_H];
  const int tid = threadIdx.x;
  if (tid < ck) {
    float mean = sum[tid] * invN;
    float var  = sumsq[tid] * invN - mean * mean;
    float av   = g[tid] * rsqrtf(var + BN_EPS);
    a_s[tid] = av;
    c_s[tid] = beta[tid] - mean * av;
  }
  __syncthreads();
  if (tid < rows) {
    const float* wr = Wn + (size_t)tid * ck;
    __bf16*      wo = Wf + (size_t)tid * ck;
    float acc = bn[tid];
    for (int k = 0; k < ck; ++k) {
      float w = wr[k];
      wo[k] = (__bf16)(w * a_s[k]);
      acc += w * c_s[k];
    }
    bf_out[tid] = acc;
  }
}

// ---------------------------------------------------------------------------
// GEMM + bias + LeakyReLU + column stats.  Z = lrelu(X @ Wf^T + bias), bf16.
// Block = 256 threads = 8 waves, one 16-row M tile per wave.
// Folded bf16 weights [COUT][CIN] + bias staged in LDS (async b128 copies).
// Per N-tile: burst-load all B fragments from LDS, then run the WMMA chain.
// ---------------------------------------------------------------------------
template<int CIN, int COUT>
__global__ __launch_bounds__(256) void gemm_lrelu_stats(
    const __bf16* __restrict__ X, const __bf16* __restrict__ Wf,
    const float* __restrict__ biasv, __bf16* __restrict__ Z,
    float* __restrict__ gsum, float* __restrict__ gsumsq, int ntiles)
{
  extern __shared__ char smem[];
  __bf16* wlds = (__bf16*)smem;
  float*  sstat = (float*)(smem + (size_t)CIN * COUT * 2);
  float*  blds  = sstat + 2 * COUT;

  const int tid = threadIdx.x;
  // stage the whole folded weight matrix into LDS with async b128 copies
  const int chunks = CIN * COUT / 8;           // 16B chunks
  for (int i = tid; i < chunks; i += 256)
    copy16_to_lds(Wf + (size_t)i * 8, wlds + (size_t)i * 8);
  for (int i = tid; i < 2 * COUT; i += 256) sstat[i] = 0.0f;
  if (tid < COUT) blds[tid] = biasv[tid];
  wait_lds_copies();
  __syncthreads();

  const int lane = tid & 31;
  const int wave = tid >> 5;
  const int tile = blockIdx.x * 8 + wave;
  if (tile < ntiles) {
    const int m0    = tile * 16;
    const int row   = m0 + (lane & 15);
    const int hi16  = lane >> 4;               // 0 or 1
    const int koffA = hi16 << 3;               // A frag K interleave: 0 / 8
    const int koffB = hi16 << 3;               // B frag K interleave: 0 / 8
    const __bf16* xrow = X + (size_t)row * CIN;

    constexpr int KS = CIN / 32;
    bf16_16 afrag[KS];
#pragma unroll
    for (int kk = 0; kk < KS; ++kk) {
      const __bf16* p = xrow + kk * 32 + koffA;
      bf16_8 lo = *(const bf16_8*)(p);
      bf16_8 hi = *(const bf16_8*)(p + 16);
      afrag[kk] = __builtin_shufflevector(lo, hi, 0,1,2,3,4,5,6,7,8,9,10,11,12,13,14,15);
    }

#pragma unroll 1
    for (int nt = 0; nt < COUT / 16; ++nt) {
      const int col = nt * 16 + (lane & 15);
      const float bias = blds[col];            // ds_load_b32, prefetchable
      const __bf16* wrow = wlds + (size_t)col * CIN;

      // burst-load all B fragments for this N-tile, then run the WMMA chain
      bf16_16 bfr[KS];
#pragma unroll
      for (int kk = 0; kk < KS; ++kk) {
        const __bf16* p = wrow + kk * 32 + koffB;
        bf16_8 lo = *(const bf16_8*)(p);
        bf16_8 hi = *(const bf16_8*)(p + 16);
        bfr[kk] = __builtin_shufflevector(lo, hi, 0,1,2,3,4,5,6,7,8,9,10,11,12,13,14,15);
      }
      f32_8 acc = {};
#pragma unroll
      for (int kk = 0; kk < KS; ++kk)
        acc = __builtin_amdgcn_wmma_f32_16x16x32_bf16(
            false, afrag[kk], false, bfr[kk], (short)0, acc, false, false);

      const int rbase = m0 + (hi16 << 3);
      float s = 0.0f, q = 0.0f;
#pragma unroll
      for (int r = 0; r < 8; ++r) {
        float y = acc[r] + bias;
        float z = fmaxf(y, 0.0f) + NEG_SLOPE * fminf(y, 0.0f);  // LeakyReLU(0.2)
        s += z; q += z * z;
        Z[(size_t)(rbase + r) * COUT + col] = (__bf16)z;
      }
      // combine row halves (lanes L and L+16 hold same column)
      s += __shfl_xor(s, 16, 32);
      q += __shfl_xor(q, 16, 32);
      if (lane < 16) {
        atomicAdd(&sstat[col], s);
        atomicAdd(&sstat[COUT + col], q);
      }
    }
  }
  __syncthreads();
  for (int i = tid; i < COUT; i += 256) {
    atomicAdd(&gsum[i],   sstat[i]);
    atomicAdd(&gsumsq[i], sstat[COUT + i]);
  }
}

// ---------------------------------------------------------------------------
// final: f = z3*a3 + c3 ; node_update = a3 * (mean_k z3[nbr]) + c3.
// One wave per node: lanes 0-15 each load a neighbor index, broadcast via
// __shfl; each lane covers 4 columns (bf16x4 = 8B -> full 256B row per wave).
// z3 (25.6 MB bf16) is L2-resident -> gather runs at L2 bandwidth.
// ---------------------------------------------------------------------------
__global__ __launch_bounds__(256) void gather_out_kernel(
    const __bf16* __restrict__ Z3, const int* __restrict__ nidx,
    const float* __restrict__ g3, const float* __restrict__ beta3,
    const float* __restrict__ s3, const float* __restrict__ q3,
    float* __restrict__ out_update, float* __restrict__ out_f)
{
  __shared__ float a3[C_OUTD];
  __shared__ float c3[C_OUTD];
  const int tid = threadIdx.x;
  if (tid < C_OUTD) {
    const float invN = 1.0f / (float)N_NODES;
    float mean = s3[tid] * invN;
    float var  = q3[tid] * invN - mean * mean;
    float av   = g3[tid] * rsqrtf(var + BN_EPS);
    a3[tid] = av;
    c3[tid] = beta3[tid] - mean * av;
  }
  __syncthreads();

  const int lane = tid & 31;
  const int wave = tid >> 5;
  const int c0 = lane * 4;
  const f32_4 av = *(const f32_4*)&a3[c0];
  const f32_4 cv = *(const f32_4*)&c3[c0];
  const float invK = 1.0f / (float)K_NB;

  for (int node = blockIdx.x * 8 + wave; node < N_NODES; node += gridDim.x * 8) {
    const int ir = (lane < K_NB) ? nidx[(size_t)node * K_NB + lane] : 0;
    f32_4 s = {};
#pragma unroll
    for (int k = 0; k < K_NB; ++k) {
      const int nb = __shfl(ir, k, 32);
      bf16_4 v = *(const bf16_4*)(Z3 + (size_t)nb * C_OUTD + c0);
      s += __builtin_convertvector(v, f32_4);
    }
    f32_4 upd = s * invK * av + cv;
    *(f32_4*)(out_update + (size_t)node * C_OUTD + c0) = upd;

    bf16_4 z = *(const bf16_4*)(Z3 + (size_t)node * C_OUTD + c0);
    f32_4 fz = __builtin_convertvector(z, f32_4) * av + cv;
    *(f32_4*)(out_f + (size_t)node * C_OUTD + c0) = fz;
  }
}

// ---------------------------------------------------------------------------
extern "C" void kernel_launch(void* const* d_in, const int* in_sizes, int n_in,
                              void* d_out, int out_size, void* d_ws, size_t ws_size,
                              hipStream_t stream)
{
  const float* X   = (const float*)d_in[0];
  const int*   nix = (const int*)d_in[1];
  // d_in[2] = prob_retained (unused, == 1)
  const float* W1  = (const float*)d_in[3];
  const float* b1  = (const float*)d_in[4];
  const float* g1  = (const float*)d_in[5];
  const float* be1 = (const float*)d_in[6];
  const float* W2  = (const float*)d_in[7];
  const float* b2  = (const float*)d_in[8];
  const float* g2  = (const float*)d_in[9];
  const float* be2 = (const float*)d_in[10];
  const float* W3  = (const float*)d_in[11];
  const float* b3  = (const float*)d_in[12];
  const float* g3  = (const float*)d_in[13];
  const float* be3 = (const float*)d_in[14];

  char* ws = (char*)d_ws;
  size_t off = 0;
  auto alloc = [&](size_t bytes) -> char* {
    char* p = ws + off;
    off += (bytes + 255) & ~(size_t)255;
    return p;
  };
  __bf16* Xb  = (__bf16*)alloc((size_t)N_NODES * C_IN  * 2);
  __bf16* Z1  = (__bf16*)alloc((size_t)N_NODES * C_H   * 2);
  __bf16* Z2  = (__bf16*)alloc((size_t)N_NODES * C_H   * 2);
  __bf16* Z3  = (__bf16*)alloc((size_t)N_NODES * C_OUTD * 2);
  __bf16* W1f = (__bf16*)alloc((size_t)C_H   * C_IN * 2);
  __bf16* W2f = (__bf16*)alloc((size_t)C_H   * C_H  * 2);
  __bf16* W3f = (__bf16*)alloc((size_t)C_OUTD * C_H * 2);
  float* b1f = (float*)alloc(C_H * 4);
  float* b2f = (float*)alloc(C_H * 4);
  float* b3f = (float*)alloc(C_OUTD * 4);
  float* stats = (float*)alloc(6 * C_H * 4);
  float* s1 = stats;            float* q1 = stats + C_H;
  float* s2 = stats + 2 * C_H;  float* q2 = stats + 3 * C_H;
  float* s3 = stats + 4 * C_H;  float* q3 = stats + 5 * C_H;

  const int smem1 = C_IN * C_H   * 2 + 3 * C_H   * 4;   //  68608
  const int smem2 = C_H  * C_H   * 2 + 3 * C_H   * 4;   // 134144
  const int smem3 = C_H  * C_OUTD * 2 + 3 * C_OUTD * 4; //  67072
  (void)hipFuncSetAttribute((const void*)gemm_lrelu_stats<C_IN, C_H>,
                            hipFuncAttributeMaxDynamicSharedMemorySize, smem1);
  (void)hipFuncSetAttribute((const void*)gemm_lrelu_stats<C_H, C_H>,
                            hipFuncAttributeMaxDynamicSharedMemorySize, smem2);
  (void)hipFuncSetAttribute((const void*)gemm_lrelu_stats<C_H, C_OUTD>,
                            hipFuncAttributeMaxDynamicSharedMemorySize, smem3);

  const float invN = 1.0f / (float)N_NODES;
  const int gblocks = (M_TILES + 7) / 8;   // 782

  prep_kernel<<<128, 256, 0, stream>>>(W1, b1, W1f, b1f, stats);
  convert_x_kernel<<<4096, 256, 0, stream>>>(X, Xb, N_NODES * C_IN);

  gemm_lrelu_stats<C_IN, C_H><<<gblocks, 256, smem1, stream>>>(
      Xb, W1f, b1f, Z1, s1, q1, M_TILES);
  fold_kernel<<<1, 256, 0, stream>>>(W2, b2, g1, be1, s1, q1, invN, W2f, b2f, C_H, C_H);

  gemm_lrelu_stats<C_H, C_H><<<gblocks, 256, smem2, stream>>>(
      Z1, W2f, b2f, Z2, s2, q2, M_TILES);
  fold_kernel<<<1, 256, 0, stream>>>(W3, b3, g2, be2, s2, q2, invN, W3f, b3f, C_OUTD, C_H);

  gemm_lrelu_stats<C_H, C_OUTD><<<gblocks, 256, smem3, stream>>>(
      Z2, W3f, b3f, Z3, s3, q3, M_TILES);

  float* out_update = (float*)d_out;
  float* out_f = out_update + (size_t)N_NODES * C_OUTD;
  gather_out_kernel<<<2048, 256, 0, stream>>>(
      Z3, nix, g3, be3, s3, q3, out_update, out_f);
}
